// GCN_57440892617236
// MI455X (gfx1250) — matrix operand order repaired and checked by reading, not compiled, not run on hardware
//
#include <hip/hip_runtime.h>

#define FIN 128
#define HID 64

typedef __attribute__((ext_vector_type(16))) _Float16 v16h;
typedef __attribute__((ext_vector_type(8)))  float    v8f;

// ---- degree: deg[dst] += 1 per edge --------------------------------------
__global__ void __launch_bounds__(256)
gcn_deg_kernel(const int* __restrict__ dst, float* __restrict__ deg, int E) {
    int e = blockIdx.x * blockDim.x + threadIdx.x;
    if (e < E) atomicAdd(&deg[dst[e]], 1.0f);
}

// ---- dinv = 1/sqrt(deg + 1) in place -------------------------------------
__global__ void __launch_bounds__(256)
gcn_dinv_kernel(float* __restrict__ deg, int N) {
    int i = blockIdx.x * blockDim.x + threadIdx.x;
    if (i < N) deg[i] = rsqrtf(deg[i] + 1.0f);
}

// ---- C[M,64] = A[M,K] @ W[K,64] via v_wmma_f32_16x16x32_f16 --------------
// One wave owns a 16x64 row-block (4 accumulators). W is staged once per
// block into LDS, f16-converted and pre-swizzled into the per-lane WMMA
// B-fragment layout: sW[ks][tn][lane][j], j -> k = ks*32 + (lane/16)*16 + j,
// col n = tn*16 + lane%16. Inner loop: A frag from global (contiguous),
// B frags as 32B LDS reads, 4 WMMAs per K-step. M % 16 == 0.
template <int K>
__global__ void __launch_bounds__(256)
gcn_gemm_wmma(const float* __restrict__ A, const float* __restrict__ W,
              float* __restrict__ C, int M) {
    constexpr int KS = K / 32;
    __shared__ alignas(32) _Float16 sW[KS * 4 * 32 * 16];

    const int tid = threadIdx.x;
    // cooperative swizzled fill of W (f32 -> f16)
    for (int idx = tid; idx < KS * 4 * 32 * 16; idx += 256) {
        int j    = idx & 15;
        int lane = (idx >> 4) & 31;
        int tn   = (idx >> 9) & 3;
        int ks   = idx >> 11;
        int k = ks * 32 + (lane >> 4) * 16 + j;
        int n = tn * 16 + (lane & 15);
        sW[idx] = (_Float16)W[k * HID + n];
    }
    __syncthreads();

    const int lane  = tid & 31;
    const int half  = lane >> 4;   // 0 or 1
    const int l16   = lane & 15;
    const int tileM = blockIdx.x * 8 + (tid >> 5);
    if (tileM * 16 >= M) return;
    const int m = tileM * 16 + l16;

    v8f acc[4] = {};
#pragma unroll
    for (int ks = 0; ks < KS; ++ks) {
        // A fragment: element i -> k = ks*32 + half*16 + i
        v16h a;
        const float* arow = A + (size_t)m * K + ks * 32 + half * 16;
#pragma unroll
        for (int i = 0; i < 8; ++i) a[i]     = (_Float16)arow[i];
#pragma unroll
        for (int i = 0; i < 8; ++i) a[8 + i] = (_Float16)arow[8 + i];

        const _Float16* bp = &sW[ks * 2048 + lane * 16];
#pragma unroll
        for (int tn = 0; tn < 4; ++tn) {
            v16h b = *(const v16h*)(bp + tn * 512);
            acc[tn] = __builtin_amdgcn_wmma_f32_16x16x32_f16(
                /*neg_a=*/false, a, /*neg_b=*/false, b,
                /*c_mod=*/(short)0, acc[tn], /*reuse_a=*/false, /*reuse_b=*/false);
        }
    }
    // D: VGPR r -> row tileM*16 + half*8 + r, col tn*16 + l16
    float* cbase = C + (size_t)(tileM * 16 + half * 8) * HID + l16;
#pragma unroll
    for (int tn = 0; tn < 4; ++tn)
#pragma unroll
        for (int r = 0; r < 8; ++r)
            cbase[(size_t)r * HID + tn * 16] = acc[tn][r];
}

// ---- edge scatter: agg[dst] += h[src] * dinv[src]*dinv[dst] --------------
// 16 threads per edge; each handles 4 consecutive features (float4 gather).
__global__ void __launch_bounds__(256)
gcn_scatter_kernel(const float* __restrict__ h, const float* __restrict__ dinv,
                   const int* __restrict__ src, const int* __restrict__ dst,
                   float* __restrict__ agg, int E) {
    int t = blockIdx.x * blockDim.x + threadIdx.x;
    int e = t >> 4;
    if (e >= E) return;
    int c = (t & 15) * 4;
    int s = src[e], d = dst[e];
    float norm = dinv[s] * dinv[d];
    const float4 hv = *(const float4*)(h + (size_t)s * HID + c);
    float* out = agg + (size_t)d * HID + c;
    atomicAdd(out + 0, hv.x * norm);
    atomicAdd(out + 1, hv.y * norm);
    atomicAdd(out + 2, hv.z * norm);
    atomicAdd(out + 3, hv.w * norm);
}

// ---- fuse self-loop + bias (+ optional ReLU), float4-vectorized ----------
__global__ void __launch_bounds__(256)
gcn_finish_kernel(float* __restrict__ agg, const float* __restrict__ h,
                  const float* __restrict__ dinv, const float* __restrict__ bias,
                  int N, int relu) {
    int t = blockIdx.x * blockDim.x + threadIdx.x;
    if (t >= N * (HID / 4)) return;
    int i = t >> 4;          // node
    int c = (t & 15) * 4;    // feature chunk
    float d  = dinv[i];
    float dd = d * d;
    float4 av = *(float4*)(agg + (size_t)i * HID + c);
    const float4 hv = *(const float4*)(h + (size_t)i * HID + c);
    const float4 bv = *(const float4*)(bias + c);
    float4 v;
    v.x = av.x + hv.x * dd + bv.x;
    v.y = av.y + hv.y * dd + bv.y;
    v.z = av.z + hv.z * dd + bv.z;
    v.w = av.w + hv.w * dd + bv.w;
    if (relu) {
        v.x = fmaxf(v.x, 0.0f); v.y = fmaxf(v.y, 0.0f);
        v.z = fmaxf(v.z, 0.0f); v.w = fmaxf(v.w, 0.0f);
    }
    *(float4*)(agg + (size_t)i * HID + c) = v;
}

// ---- per-graph node counts -----------------------------------------------
__global__ void __launch_bounds__(256)
gcn_count_kernel(const int* __restrict__ batch, float* __restrict__ gcnt, int N) {
    int i = blockIdx.x * blockDim.x + threadIdx.x;
    if (i < N) atomicAdd(&gcnt[batch[i]], 1.0f);
}

// ---- pooled sums: gsum[batch[i]] += h2[i] --------------------------------
__global__ void __launch_bounds__(256)
gcn_poolsum_kernel(const float* __restrict__ h, const int* __restrict__ batch,
                   float* __restrict__ gsum, int N) {
    int t = blockIdx.x * blockDim.x + threadIdx.x;
    int i = t >> 4;
    if (i >= N) return;
    int c = (t & 15) * 4;
    int b = batch[i];
    const float4 hv = *(const float4*)(h + (size_t)i * HID + c);
    float* out = gsum + (size_t)b * HID + c;
    atomicAdd(out + 0, hv.x);
    atomicAdd(out + 1, hv.y);
    atomicAdd(out + 2, hv.z);
    atomicAdd(out + 3, hv.w);
}

// ---- head: out[g] = mean(gsum[g]) . Wl + bl ------------------------------
__global__ void __launch_bounds__(256)
gcn_final_kernel(const float* __restrict__ gsum, const float* __restrict__ gcnt,
                 const float* __restrict__ Wl, const float* __restrict__ bl,
                 float* __restrict__ out, int B) {
    int g = blockIdx.x * blockDim.x + threadIdx.x;
    if (g >= B) return;
    float inv = 1.0f / fmaxf(gcnt[g], 1.0f);
    float acc = bl[0];
#pragma unroll 8
    for (int f = 0; f < HID; ++f) acc += gsum[(size_t)g * HID + f] * inv * Wl[f];
    out[g] = acc;
}

extern "C" void kernel_launch(void* const* d_in, const int* in_sizes, int n_in,
                              void* d_out, int out_size, void* d_ws, size_t ws_size,
                              hipStream_t stream) {
    (void)n_in; (void)ws_size;
    const float* x     = (const float*)d_in[0];
    const int*   ei    = (const int*)d_in[1];
    const int*   batch = (const int*)d_in[2];
    // d_in[3] = batch_size scalar; B == out_size
    const float* W1 = (const float*)d_in[4];
    const float* b1 = (const float*)d_in[5];
    const float* W2 = (const float*)d_in[6];
    const float* b2 = (const float*)d_in[7];
    const float* Wl = (const float*)d_in[8];
    const float* bl = (const float*)d_in[9];
    float* out = (float*)d_out;

    const int N = in_sizes[0] / FIN;
    const int E = in_sizes[1] / 2;
    const int B = out_size;
    const int* src = ei;
    const int* dst = ei + E;

    // Workspace (floats): deg/dinv[N] | bufA[N*64] | bufB[N*64] | gsum[B*64] | gcnt[B]
    float* ws   = (float*)d_ws;
    float* deg  = ws;
    float* bufA = deg + N;
    float* bufB = bufA + (size_t)N * HID;
    float* gsum = bufB + (size_t)N * HID;
    float* gcnt = gsum + (size_t)B * HID;

    hipMemsetAsync(deg,  0, (size_t)N * sizeof(float), stream);
    hipMemsetAsync(bufB, 0, (size_t)N * HID * sizeof(float), stream);
    hipMemsetAsync(gsum, 0, (size_t)(B * HID + B) * sizeof(float), stream);

    // degrees & normalization
    gcn_deg_kernel <<<(E + 255) / 256, 256, 0, stream>>>(dst, deg, E);
    gcn_dinv_kernel<<<(N + 255) / 256, 256, 0, stream>>>(deg, N);

    const int rowTiles = (N + 15) / 16;          // 16-row blocks
    const int gBlks    = (rowTiles + 7) / 8;     // 8 waves (row-blocks) / block
    const long long st = (long long)E * 16;
    const int sBlks    = (int)((st + 255) / 256);
    const int fBlks    = (N * (HID / 4) + 255) / 256;

    // Layer 1: h = x @ W1 -> bufA; agg -> bufB; bufB = relu(agg + h*d^2 + b1)
    gcn_gemm_wmma<FIN><<<gBlks, 256, 0, stream>>>(x, W1, bufA, N);
    gcn_scatter_kernel<<<sBlks, 256, 0, stream>>>(bufA, deg, src, dst, bufB, E);
    gcn_finish_kernel <<<fBlks, 256, 0, stream>>>(bufB, bufA, deg, b1, N, 1);

    // Layer 2: h2 = bufB @ W2 -> bufA; zero bufB; agg -> bufB; finish (no relu)
    gcn_gemm_wmma<HID><<<gBlks, 256, 0, stream>>>(bufB, W2, bufA, N);
    hipMemsetAsync(bufB, 0, (size_t)N * HID * sizeof(float), stream);
    gcn_scatter_kernel<<<sBlks, 256, 0, stream>>>(bufA, deg, src, dst, bufB, E);
    gcn_finish_kernel <<<fBlks, 256, 0, stream>>>(bufB, bufA, deg, b2, N, 0);

    // Pool + head
    gcn_count_kernel  <<<(N + 255) / 256, 256, 0, stream>>>(batch, gcnt, N);
    const long long pt = (long long)N * 16;
    gcn_poolsum_kernel<<<(int)((pt + 255) / 256), 256, 0, stream>>>(bufB, batch, gsum, N);
    gcn_final_kernel  <<<(B + 255) / 256, 256, 0, stream>>>(gsum, gcnt, Wl, bl, out, B);
}
